// ScaledDotProductAttention_51891794870473
// MI455X (gfx1250) — matrix operand order repaired and checked by reading, not compiled, not run on hardware
//
#include <hip/hip_runtime.h>
#include <hip/hip_bf16.h>
#include <math.h>

typedef __bf16 bf16;
typedef __attribute__((ext_vector_type(16))) __bf16 v16bf;
typedef __attribute__((ext_vector_type(8)))  __bf16 v8bf;
typedef __attribute__((ext_vector_type(8)))  float  v8f;

#define D_MODEL 1024
#define SEQ     4096
#define BATCH   4
#define SCALE   0.125f        // 1/sqrt(64), per reference quirk
#define NEG_INF -3.0e38f

// ---------------------------------------------------------------------------
// helpers
// ---------------------------------------------------------------------------
__device__ __forceinline__ v8bf cvt8(v8f v) {
  v8bf r;
#pragma unroll
  for (int i = 0; i < 8; ++i) r[i] = (__bf16)v[i];
  return r;
}

__device__ __forceinline__ v16bf cvt16(v8f lo, v8f hi) {
  v16bf r;
#pragma unroll
  for (int i = 0; i < 8; ++i) { r[i] = (__bf16)lo[i]; r[i + 8] = (__bf16)hi[i]; }
  return r;
}

__device__ __forceinline__ v16bf cat8(v8bf lo, v8bf hi) {
  v16bf r;
#pragma unroll
  for (int i = 0; i < 8; ++i) { r[i] = lo[i]; r[i + 8] = hi[i]; }
  return r;
}

// ---------------------------------------------------------------------------
// Phase 1: QKV projection.  C[m][n] = sum_d x[m][d] * W[n][d] + b[n]
// WG = 8 waves; x row-tile converted to bf16 in LDS once, shared by all waves.
// Two independent WMMA accumulators break the D->C hazard chain.
// Output bf16, row-major for Q/K, transposed [B][D][S] for V.
// ---------------------------------------------------------------------------
__global__ __launch_bounds__(256)
void qkv_proj_kernel(const float* __restrict__ x,
                     const float* __restrict__ W,
                     const float* __restrict__ bias,
                     bf16* __restrict__ out,
                     int transpose_v)
{
  __shared__ __attribute__((aligned(64))) bf16 xl[16][D_MODEL];  // 32KB

  const int tid   = threadIdx.x;
  const int lane  = tid & 31;
  const int wave  = tid >> 5;                // 0..7
  const int half  = lane >> 4;               // 0|1
  const int lan16 = lane & 15;

  const int m0 = blockIdx.x * 16;            // flat row in [0, B*S)
  const int n0 = blockIdx.y * 128 + wave * 16;

  // --- cooperative fp32->bf16 conversion of the shared x tile (once) ---
  {
    const float* xbase = x + (size_t)m0 * D_MODEL;
    const int e0 = tid * 64;                 // 16*1024/256 = 64 elems/thread
#pragma unroll
    for (int u = 0; u < 8; ++u) {
      v8f v = *(const v8f*)(xbase + e0 + u * 8);
      *(v8bf*)&(((bf16*)xl)[e0 + u * 8]) = cvt8(v);
    }
  }
  __syncthreads();

  const bf16*  xrow = &xl[lan16][0];
  const float* wrow = W + (size_t)(n0 + lan16) * D_MODEL;

  v8f acc0 = {}, acc1 = {};
#pragma unroll 2
  for (int k0 = 0; k0 < D_MODEL; k0 += 64) {
    // ---- K-step k0 -> acc0 ----
    {
      const int aoff = k0 + (half ? 16 : 0);
      v16bf af = cat8(*(const v8bf*)(xrow + aoff),
                      *(const v8bf*)(xrow + aoff + 8));
      const int boff = k0 + half * 16;
      v16bf bf = cvt16(*(const v8f*)(wrow + boff),
                       *(const v8f*)(wrow + boff + 8));
      acc0 = __builtin_amdgcn_wmma_f32_16x16x32_bf16(false, af, false, bf,
                                                     (short)0, acc0, false, false);
    }
    // ---- K-step k0+32 -> acc1 ----
    {
      const int aoff = k0 + 32 + (half ? 16 : 0);
      v16bf af = cat8(*(const v8bf*)(xrow + aoff),
                      *(const v8bf*)(xrow + aoff + 8));
      const int boff = k0 + 32 + half * 16;
      v16bf bf = cvt16(*(const v8f*)(wrow + boff),
                       *(const v8f*)(wrow + boff + 8));
      acc1 = __builtin_amdgcn_wmma_f32_16x16x32_bf16(false, af, false, bf,
                                                     (short)0, acc1, false, false);
    }
  }

  const float bval = bias[n0 + lan16];
  v8f acc;
#pragma unroll
  for (int r = 0; r < 8; ++r) acc[r] = acc0[r] + acc1[r] + bval;

  if (!transpose_v) {
#pragma unroll
    for (int r = 0; r < 8; ++r) {
      const int row = m0 + r + 8 * half;     // C layout: M = r + 8*(lane/16)
      out[(size_t)row * D_MODEL + n0 + lan16] = (bf16)acc[r];
    }
  } else {
#pragma unroll
    for (int r = 0; r < 8; ++r) {
      const int row = m0 + r + 8 * half;     // flat b*S + s
      const int bb  = row >> 12;             // / SEQ
      const int ss  = row & (SEQ - 1);
      out[((size_t)bb * D_MODEL + n0 + lan16) * SEQ + ss] = (bf16)acc[r];
    }
  }
}

// ---------------------------------------------------------------------------
// Phase 2: causal flash attention, single head over full D=1024.
// WG = 16 query rows, 8 waves; wave w owns D-slice [128w, 128w+128).
// 4 independent score accumulators; fully parallel shuffle-based softmax.
// ---------------------------------------------------------------------------
__global__ __launch_bounds__(256)
void flash_attn_kernel(const bf16* __restrict__ Q,
                       const bf16* __restrict__ K,
                       const bf16* __restrict__ Vt,
                       float* __restrict__ out)
{
  __shared__ float red[8][16][32];                          // per-wave score partials
  __shared__ __attribute__((aligned(64))) bf16 pbf[16][32]; // probabilities (bf16)
  __shared__ __attribute__((aligned(32))) float row_m[16];
  __shared__ __attribute__((aligned(32))) float row_l[16];
  __shared__ __attribute__((aligned(32))) float row_alpha[16];

  const int tid   = threadIdx.x;
  const int lane  = tid & 31;
  const int wave  = tid >> 5;
  const int half  = lane >> 4;
  const int lan16 = lane & 15;

  const int b     = blockIdx.y;
  const int qbase = blockIdx.x * 16;
  const int d0    = wave * 128;

  const bf16* Qb = Q  + ((size_t)b * SEQ + qbase) * D_MODEL;
  const bf16* Kb = K  + (size_t)b * SEQ * D_MODEL;
  const bf16* Vb = Vt + (size_t)b * D_MODEL * SEQ;

  // Preload Q A-fragments for this wave's D-slice (4 x 16x32)
  v16bf qf[4];
#pragma unroll
  for (int t = 0; t < 4; ++t) {
    const bf16* qr = Qb + (size_t)lan16 * D_MODEL + d0 + t * 32 + (half ? 16 : 0);
    qf[t] = cat8(*(const v8bf*)qr, *(const v8bf*)(qr + 8));
  }

  const v8f vzero = {};
  v8f oacc[8];
#pragma unroll
  for (int i = 0; i < 8; ++i) oacc[i] = vzero;

  if (tid < 16) { row_m[tid] = NEG_INF; row_l[tid] = 0.0f; }
  __syncthreads();

  const int jend = qbase + 15;
  for (int j = 0; j <= jend; j += 32) {
    if (j + 32 <= jend) {  // prefetch next key tile (one row per lane)
      __builtin_prefetch(Kb + (size_t)(j + 32 + lane) * D_MODEL + d0, 0, 1);
    }

    // --- 1) partial scores: 4 independent accumulators (no WMMA RAW chain) ---
    v8f s0a = vzero, s0b = vzero, s1a = vzero, s1b = vzero;
#pragma unroll
    for (int t = 0; t < 4; t += 2) {
      const int kc0 = d0 + t * 32 + half * 16;
      const int kc1 = kc0 + 32;
      v16bf b00 = *(const v16bf*)(Kb + (size_t)(j + lan16) * D_MODEL + kc0);
      v16bf b10 = *(const v16bf*)(Kb + (size_t)(j + 16 + lan16) * D_MODEL + kc0);
      v16bf b01 = *(const v16bf*)(Kb + (size_t)(j + lan16) * D_MODEL + kc1);
      v16bf b11 = *(const v16bf*)(Kb + (size_t)(j + 16 + lan16) * D_MODEL + kc1);
      s0a = __builtin_amdgcn_wmma_f32_16x16x32_bf16(false, qf[t],     false, b00,
                                                    (short)0, s0a, false, false);
      s1a = __builtin_amdgcn_wmma_f32_16x16x32_bf16(false, qf[t],     false, b10,
                                                    (short)0, s1a, false, false);
      s0b = __builtin_amdgcn_wmma_f32_16x16x32_bf16(false, qf[t + 1], false, b01,
                                                    (short)0, s0b, false, false);
      s1b = __builtin_amdgcn_wmma_f32_16x16x32_bf16(false, qf[t + 1], false, b11,
                                                    (short)0, s1b, false, false);
    }

    // --- 2) deposit partials (C layout: row = r + 8*(lane/16), col = lane%16) ---
#pragma unroll
    for (int r = 0; r < 8; ++r) {
      red[wave][r + 8 * half][lan16]      = s0a[r] + s0b[r];
      red[wave][r + 8 * half][lan16 + 16] = s1a[r] + s1b[r];
    }
    __syncthreads();

    // --- 3) cross-wave reduce + causal mask + parallel online softmax ---
    {
      const int row = tid >> 4;           // 16 lanes per row
      const int c0  = tid & 15;
      float s0 = 0.0f, s1 = 0.0f;
#pragma unroll
      for (int w = 0; w < 8; ++w) { s0 += red[w][row][c0]; s1 += red[w][row][c0 + 16]; }
      s0 *= SCALE; s1 *= SCALE;
      if (j + c0      > qbase + row) s0 = NEG_INF;
      if (j + c0 + 16 > qbase + row) s1 = NEG_INF;

      float tmax = fmaxf(s0, s1);
#pragma unroll
      for (int m = 1; m <= 8; m <<= 1)
        tmax = fmaxf(tmax, __shfl_xor(tmax, m, 32));   // stays in 16-lane half

      const float mold = row_m[row];
      const float mnew = fmaxf(mold, tmax);
      const float p0 = __expf(s0 - mnew);
      const float p1 = __expf(s1 - mnew);
      pbf[row][c0]      = (bf16)p0;
      pbf[row][c0 + 16] = (bf16)p1;

      float lsum = p0 + p1;
#pragma unroll
      for (int m = 1; m <= 8; m <<= 1)
        lsum += __shfl_xor(lsum, m, 32);

      if (c0 == 0) {
        const float alpha = __expf(mold - mnew);
        row_alpha[row] = alpha;
        row_l[row]     = row_l[row] * alpha + lsum;
        row_m[row]     = mnew;
      }
    }
    __syncthreads();

    // --- 4) rescale O, then O += P * V (8 independent WMMA accumulators) ---
    const v8f av = *(const v8f*)&row_alpha[8 * half];
#pragma unroll
    for (int i = 0; i < 8; ++i) oacc[i] *= av;

    const bf16* pr = &pbf[lan16][half ? 16 : 0];
    v16bf pf = cat8(*(const v8bf*)pr, *(const v8bf*)(pr + 8));

#pragma unroll
    for (int i = 0; i < 8; ++i) {
      const bf16* vr = Vb + (size_t)(d0 + 16 * i + lan16) * SEQ + j + half * 16;
      v16bf bvf = *(const v16bf*)vr;
      oacc[i] = __builtin_amdgcn_wmma_f32_16x16x32_bf16(false, pf, false, bvf,
                                                        (short)0, oacc[i], false, false);
    }
    __syncthreads();
  }

  // --- normalize by l and write fp32 output ---
  const v8f lv = *(const v8f*)&row_l[8 * half];
  v8f inv;
#pragma unroll
  for (int r = 0; r < 8; ++r) inv[r] = 1.0f / lv[r];

  float* ob = out + ((size_t)b * SEQ + qbase) * D_MODEL;
#pragma unroll
  for (int i = 0; i < 8; ++i) {
    const v8f o = oacc[i] * inv;
#pragma unroll
    for (int r = 0; r < 8; ++r) {
      ob[(size_t)(r + 8 * half) * D_MODEL + d0 + 16 * i + lan16] = o[r];
    }
  }
}

// ---------------------------------------------------------------------------
// launch
// ---------------------------------------------------------------------------
extern "C" void kernel_launch(void* const* d_in, const int* in_sizes, int n_in,
                              void* d_out, int out_size, void* d_ws, size_t ws_size,
                              hipStream_t stream) {
  (void)in_sizes; (void)n_in; (void)out_size; (void)ws_size;

  const float* x  = (const float*)d_in[0];
  const float* Wq = (const float*)d_in[1];
  const float* bq = (const float*)d_in[2];
  const float* Wk = (const float*)d_in[3];
  const float* bk = (const float*)d_in[4];
  const float* Wv = (const float*)d_in[5];
  const float* bv = (const float*)d_in[6];
  float* out = (float*)d_out;

  const size_t elems = (size_t)BATCH * SEQ * D_MODEL;
  bf16* Qb = (bf16*)d_ws;          // [B][S][D] bf16
  bf16* Kb = Qb + elems;           // [B][S][D] bf16
  bf16* Vt = Kb + elems;           // [B][D][S] bf16 (transposed)

  dim3 gp(BATCH * SEQ / 16, D_MODEL / 128, 1);
  qkv_proj_kernel<<<gp, 256, 0, stream>>>(x, Wq, bq, Qb, 0);
  qkv_proj_kernel<<<gp, 256, 0, stream>>>(x, Wk, bk, Kb, 0);
  qkv_proj_kernel<<<gp, 256, 0, stream>>>(x, Wv, bv, Vt, 1);

  dim3 ga(SEQ / 16, BATCH, 1);
  flash_attn_kernel<<<ga, 256, 0, stream>>>(Qb, Kb, Vt, out);
}
